// ConvSelfAttention_85263690760236
// MI455X (gfx1250) — compile-verified, matrix operand order checked by reading, not run on hardware
//
#include <hip/hip_runtime.h>
#include <hip/hip_bf16.h>

typedef __attribute__((ext_vector_type(16))) __bf16 v16bf;
typedef __attribute__((ext_vector_type(8)))  float  v8f;
typedef unsigned int u32;

// ---------- helpers ----------
__device__ __host__ inline __bf16 f2bf(float f) {
  u32 u = __builtin_bit_cast(u32, f);
  u += 0x7fffu + ((u >> 16) & 1u);           // round-to-nearest-even
  unsigned short h = (unsigned short)(u >> 16);
  return __builtin_bit_cast(__bf16, h);
}

// K index inside a 16-bit WMMA A/B fragment for element e (0..15) given the
// lane half (lane<16 -> 0, lane>=16 -> 1).  Matches ISA 7.12.2.
__device__ inline int kmap(int e, int half) {
  int j = e >> 1, p = e & 1;
  int base = (j < 4) ? 0 : 16;
  return base + half * 8 + ((j & 3) << 1) + p;
}

__device__ inline v8f wmma_bf16(v16bf a, v16bf b, v8f c) {
  return __builtin_amdgcn_wmma_f32_16x16x32_bf16(false, a, false, b, (short)0, c,
                                                 false, false);
}

// async global -> LDS copy of 16 bytes (gfx1250, ASYNCcnt-tracked)
__device__ inline void async_copy_b128(u32 lds_off, const __bf16* gaddr) {
  asm volatile("global_load_async_to_lds_b128 %0, %1, off"
               :: "v"(lds_off), "v"(gaddr)
               : "memory");
}
__device__ inline void wait_async0() {
  asm volatile("s_wait_asynccnt 0x0" ::: "memory");
}
__device__ inline u32 lds_off_of(const void* p) {
  return (u32)(unsigned long long)p;   // low 32 bits of generic = LDS offset
}

// ---------- layout conversion kernels ----------
// inp (q, ic, p) f32 -> xb[(q*64+p)*512 + ic] bf16
__global__ __launch_bounds__(256) void cvt_x_kernel(const float* __restrict__ in,
                                                    __bf16* __restrict__ out) {
  size_t i = (size_t)blockIdx.x * 256 + threadIdx.x;
  const size_t n = (size_t)384 * 64 * 512;
  if (i >= n) return;
  int ic = (int)(i % 512);
  size_t t = i / 512;
  int p = (int)(t % 64);
  int q = (int)(t / 64);
  out[i] = f2bf(in[((size_t)q * 512 + ic) * 64 + p]);
}

// w (oc, ic, 3, 3) f32 -> wb[((tap*16+ks)*NOC + oc)*32 + k] bf16  (ic = ks*32+k)
// i.e. K-slice-major, then oc, then the 32 K values contiguous -> B fragments
// become two contiguous 16-byte groups per lane.
template <int NOC>
__global__ __launch_bounds__(256) void cvt_w_kernel(const float* __restrict__ in,
                                                    __bf16* __restrict__ out) {
  size_t i = (size_t)blockIdx.x * 256 + threadIdx.x;
  const size_t n = (size_t)9 * 512 * NOC;
  if (i >= n) return;
  int k = (int)(i & 31);
  size_t t = i >> 5;
  int oc = (int)(t % NOC);
  size_t s = t / NOC;
  int ks = (int)(s % 16);
  int tap = (int)(s / 16);
  int ic = ks * 32 + k;
  out[i] = f2bf(in[((size_t)oc * 512 + ic) * 9 + tap]);
}

// ---------- 3x3 conv as implicit GEMM with WMMA ----------
// One workgroup: image q x 256 out-channel tile.  M = 64 pixels, K = 9*512.
// 8 waves in a 4(M) x 2(N) grid, each wave: 16 rows x 128 cols = 8 WMMA tiles.
// Weights double-buffered in LDS via async global->LDS DMA (ASYNCcnt).
// EPI==0: conv_in epilogue -> feats (embs 0-3: [b][q][d] bf16, emb 4 (V): [b][d][q])
// EPI==1: conv_out epilogue -> out[q][oc][px] f32 (+bias)
template <int NOC, int EPI>
__global__ __launch_bounds__(256) void conv3x3_wmma_kernel(
    const __bf16* __restrict__ X, const __bf16* __restrict__ W,
    const float* __restrict__ bias, __bf16* __restrict__ feats,
    float* __restrict__ outp) {
  __shared__ __bf16 sX[64 * 512];        // whole-image input tile (64 KB)
  __shared__ __bf16 sW[2][256 * 32];     // double-buffered K-slice (2 x 16 KB)

  const int q   = blockIdx.x;
  const int ocb = blockIdx.y * 256;
  const int tid = threadIdx.x;
  const int wave = tid >> 5, lane = tid & 31;
  const int wm = wave & 3, wn = wave >> 2;
  const int l15 = lane & 15, half = lane >> 4;
  const int NS = 144;                    // 9 taps * 16 K-slices

  // ---- async-stage the input tile (64 KB, 16 b128 chunks per thread)
  {
    const __bf16* src = X + (size_t)q * 64 * 512;
    const u32 lbase = lds_off_of(&sX[0]);
#pragma unroll
    for (int j = 0; j < 16; ++j) {
      const int c = tid + j * 256;       // 0..4095 b128 chunks
      async_copy_b128(lbase + c * 16, src + c * 8);
    }
  }
  // ---- async-stage weight slice 0 into buffer 0 (16 KB, 4 chunks per thread)
  {
    const __bf16* src = W + ((size_t)0 * NOC + ocb) * 32;
    const u32 lbase = lds_off_of(&sW[0][0]);
#pragma unroll
    for (int j = 0; j < 4; ++j) {
      const int c = tid + j * 256;       // 0..1023
      async_copy_b128(lbase + c * 16, src + c * 8);
    }
  }

  v8f acc[8] = {};
  const int arow_m = wm * 16 + l15;      // A-fragment row = pixel
  const int py = arow_m >> 3, px = arow_m & 7;

  for (int step = 0; step < NS; ++step) {
    wait_async0();      // our DMA for slice `step` (and X on step 0) landed
    __syncthreads();    // everyone waited + finished compute of step-1
    if (step + 1 < NS) {  // prefetch next slice into the other buffer
      const __bf16* src = W + ((size_t)(step + 1) * NOC + ocb) * 32;
      const u32 lbase = lds_off_of(&sW[(step + 1) & 1][0]);
#pragma unroll
      for (int j = 0; j < 4; ++j) {
        const int c = tid + j * 256;
        async_copy_b128(lbase + c * 16, src + c * 8);
      }
    }

    const int tap = step >> 4, ks = step & 15;
    const int dy = tap / 3 - 1, dx = tap % 3 - 1;
    const int sy = py + dy, sx = px + dx;
    const bool valid = ((unsigned)sy < 8u) && ((unsigned)sx < 8u);

    v16bf a;
    if (valid) {
      const __bf16* arow = sX + (sy * 8 + sx) * 512 + ks * 32;
#pragma unroll
      for (int e = 0; e < 16; ++e) a[e] = arow[kmap(e, half)];
    } else {
      const __bf16 z = __builtin_bit_cast(__bf16, (unsigned short)0);
#pragma unroll
      for (int e = 0; e < 16; ++e) a[e] = z;
    }

    const __bf16* wbuf = &sW[step & 1][0];
#pragma unroll
    for (int nt = 0; nt < 8; ++nt) {
      const __bf16* wrow = wbuf + (wn * 128 + nt * 16 + l15) * 32;
      v16bf b;
#pragma unroll
      for (int e = 0; e < 16; ++e) b[e] = wrow[kmap(e, half)];
      acc[nt] = wmma_bf16(a, b, acc[nt]);
    }
  }

  // epilogue: C/D layout — lane%16 = N col, VGPR r -> M = r + 8*(lane/16)
#pragma unroll
  for (int nt = 0; nt < 8; ++nt) {
    const int oc = ocb + wn * 128 + nt * 16 + l15;
    const float bv = bias[oc];
#pragma unroll
    for (int r = 0; r < 8; ++r) {
      const int pix = wm * 16 + r + 8 * half;
      const float v = acc[nt][r] + bv;
      if (EPI == 0) {
        // oc = (emb*64+d)*8 + head
        const int head = oc & 7, t2 = oc >> 3, emb = t2 >> 6, d = t2 & 63;
        const size_t base = (size_t)(emb * 512 + head * 64 + pix) * 24576;
        if (emb == 4)  // V stored transposed: [batch][d][q]
          feats[base + (size_t)d * 384 + q] = f2bf(v);
        else           // Q/K stored [batch][q][d]
          feats[base + (size_t)q * 64 + d] = f2bf(v);
      } else {
        outp[((size_t)q * NOC + oc) * 64 + pix] = v;
      }
    }
  }
}

// ---------- agent-aware attention with WMMA ----------
// grid (512 batches, 12 q-tiles of 32), block 256 (8 waves).
// feats: embs 0-3 [emb][batch][q][d]; emb 4 (V) transposed [batch][d][q].
__global__ __launch_bounds__(256) void attn_wmma_kernel(
    const __bf16* __restrict__ feats, const float* __restrict__ attn_mask,
    const int* __restrict__ agent, __bf16* __restrict__ xb2) {
  __shared__ float  sS[2][32][384];   // 96 KB: scores same/other
  __shared__ __bf16 sP[32][384];      // 24 KB: combined probs (bf16)
  __shared__ __bf16 sQ[2][32][64];    // 8 KB: q_same/q_other tiles

  const int batch = blockIdx.x;           // head*64 + pix
  const int head = batch >> 6, pix = batch & 63;
  const int q0 = blockIdx.y * 32;
  const int tid = threadIdx.x;
  const int wave = tid >> 5, lane = tid & 31;
  const int l15 = lane & 15, half = lane >> 4;

  const size_t bstride = (size_t)384 * 64;
  const __bf16* f_ks = feats + (size_t)(0 * 512 + batch) * bstride;
  const __bf16* f_ko = feats + (size_t)(1 * 512 + batch) * bstride;
  const __bf16* f_qs = feats + (size_t)(2 * 512 + batch) * bstride;
  const __bf16* f_qo = feats + (size_t)(3 * 512 + batch) * bstride;
  const __bf16* f_v  = feats + (size_t)(4 * 512 + batch) * bstride;  // [d][q]

  // stage Q tiles
  {
    const u32* s0 = (const u32*)(f_qs + (size_t)q0 * 64);
    const u32* s1 = (const u32*)(f_qo + (size_t)q0 * 64);
    u32* d0 = (u32*)&sQ[0][0][0];
    u32* d1 = (u32*)&sQ[1][0][0];
    for (int i = tid; i < 32 * 64 / 2; i += 256) { d0[i] = s0[i]; d1[i] = s1[i]; }
  }
  __syncthreads();

  // ---- S = (Q*scale) @ K^T + mask : 2 mats x 2 Mtiles x 24 Ntiles = 96 tiles
  for (int i = 0; i < 12; ++i) {
    const int t = wave * 12 + i;
    const int mat = t / 48, rem = t % 48, mt = rem / 24, nt = rem % 24;
    const __bf16* fk = mat ? f_ko : f_ks;
    v8f acc = {};
#pragma unroll
    for (int kk = 0; kk < 2; ++kk) {
      const __bf16* arow = &sQ[mat][mt * 16 + l15][kk * 32];
      v16bf a;
#pragma unroll
      for (int e = 0; e < 16; ++e) a[e] = arow[kmap(e, half)];
      const __bf16* brow = fk + (size_t)(nt * 16 + l15) * 64 + kk * 32;
      v16bf b;
#pragma unroll
      for (int e = 0; e < 16; ++e) b[e] = brow[kmap(e, half)];
      acc = wmma_bf16(a, b, acc);
    }
    const int col = nt * 16 + l15;
#pragma unroll
    for (int r = 0; r < 8; ++r) {
      const int m = mt * 16 + r + 8 * half;
      const int qg = q0 + m;
      sS[mat][m][col] =
          acc[r] * 0.125f + attn_mask[((size_t)head * 384 + qg) * 384 + col];
    }
  }
  __syncthreads();

  // ---- row softmax (64 rows, 8 per wave, lanes cooperate along k)
  for (int rr = 0; rr < 8; ++rr) {
    const int row = wave * 8 + rr;
    const int mat = row >> 5, m = row & 31;
    float* srow = sS[mat][m];
    float mx = -__builtin_inff();
    for (int j = lane; j < 384; j += 32) mx = fmaxf(mx, srow[j]);
#pragma unroll
    for (int off = 16; off; off >>= 1) mx = fmaxf(mx, __shfl_xor(mx, off, 32));
    float sum = 0.f;
    for (int j = lane; j < 384; j += 32) {
      const float p = __expf(srow[j] - mx);
      srow[j] = p;
      sum += p;
    }
#pragma unroll
    for (int off = 16; off; off >>= 1) sum += __shfl_xor(sum, off, 32);
    const float inv = 1.0f / sum;
    for (int j = lane; j < 384; j += 32) srow[j] *= inv;
  }
  __syncthreads();

  // ---- agent-aware combine -> bf16 P
  for (int i = tid; i < 32 * 384; i += 256) {
    const int m = i / 384, k = i % 384;
    const float mm = (float)agent[((size_t)head * 384 + q0 + m) * 384 + k];
    sP[m][k] = f2bf(mm * sS[0][m][k] + (1.0f - mm) * sS[1][m][k]);
  }
  __syncthreads();

  // ---- out = P @ V : 2 Mtiles x 4 Ntiles = 8 WMMA tiles, one per wave
  {
    const int mt = wave >> 2, nt = wave & 3;
    v8f acc = {};
    for (int kk = 0; kk < 12; ++kk) {
      const __bf16* arow = &sP[mt * 16 + l15][kk * 32];
      v16bf a;
#pragma unroll
      for (int e = 0; e < 16; ++e) a[e] = arow[kmap(e, half)];
      // V transposed: B[k=kq][n=d] = f_v[d*384 + kq] -> contiguous in kq
      const __bf16* bp = f_v + (size_t)(nt * 16 + l15) * 384 + kk * 32;
      v16bf b;
#pragma unroll
      for (int e = 0; e < 16; ++e) b[e] = bp[kmap(e, half)];
      acc = wmma_bf16(a, b, acc);
    }
#pragma unroll
    for (int r = 0; r < 8; ++r) {
      const int m = mt * 16 + r + 8 * half;
      const int qg = q0 + m;
      const int d = nt * 16 + l15;
      // conv_out input layout: xb2[(q*64+pix)*512 + d*8 + head]
      xb2[((size_t)qg * 64 + pix) * 512 + d * 8 + head] = f2bf(acc[r]);
    }
  }
}

// ---------- host launch ----------
extern "C" void kernel_launch(void* const* d_in, const int* in_sizes, int n_in,
                              void* d_out, int out_size, void* d_ws, size_t ws_size,
                              hipStream_t stream) {
  (void)in_sizes; (void)n_in; (void)out_size; (void)ws_size;
  const float* inp       = (const float*)d_in[0];
  const float* attn_mask = (const float*)d_in[1];
  const int*   agent     = (const int*)d_in[2];
  const float* w_in      = (const float*)d_in[3];
  const float* b_in      = (const float*)d_in[4];
  const float* w_out     = (const float*)d_in[5];
  const float* b_out     = (const float*)d_in[6];
  float* out = (float*)d_out;

  char* base = (char*)d_ws;
  size_t off = 0;
  auto wsalloc = [&](size_t bytes) -> char* {
    char* p = base + off;
    off += (bytes + 255) & ~(size_t)255;
    return p;
  };
  __bf16* xb    = (__bf16*)wsalloc((size_t)384 * 64 * 512 * 2);        // input bf16
  __bf16* wbin  = (__bf16*)wsalloc((size_t)9 * 512 * 2560 * 2);        // w_in bf16
  __bf16* wbout = (__bf16*)wsalloc((size_t)9 * 512 * 512 * 2);         // w_out bf16
  __bf16* feats = (__bf16*)wsalloc((size_t)5 * 512 * 384 * 64 * 2);    // conv_in out
  __bf16* xb2   = (__bf16*)wsalloc((size_t)384 * 64 * 512 * 2);        // attn out

  {
    const size_t n = (size_t)384 * 64 * 512;
    cvt_x_kernel<<<dim3((unsigned)((n + 255) / 256)), 256, 0, stream>>>(inp, xb);
  }
  {
    const size_t n = (size_t)9 * 512 * 2560;
    cvt_w_kernel<2560><<<dim3((unsigned)((n + 255) / 256)), 256, 0, stream>>>(w_in, wbin);
  }
  {
    const size_t n = (size_t)9 * 512 * 512;
    cvt_w_kernel<512><<<dim3((unsigned)((n + 255) / 256)), 256, 0, stream>>>(w_out, wbout);
  }

  conv3x3_wmma_kernel<2560, 0><<<dim3(384, 10), 256, 0, stream>>>(
      xb, wbin, b_in, feats, nullptr);

  attn_wmma_kernel<<<dim3(512, 12), 256, 0, stream>>>(feats, attn_mask, agent, xb2);

  conv3x3_wmma_kernel<512, 1><<<dim3(384, 2), 256, 0, stream>>>(
      xb2, wbout, b_out, nullptr, out);
}
